// QuantumInterferenceRouter_54013508715145
// MI455X (gfx1250) — matrix operand order repaired and checked by reading, not compiled, not run on hardware
//
#include <hip/hip_runtime.h>
#include <hip/hip_bf16.h>

// ---------- types ----------
using f16 = _Float16;
typedef __attribute__((ext_vector_type(8)))  _Float16 v8h;
typedef __attribute__((ext_vector_type(16))) _Float16 v16h;
typedef __attribute__((ext_vector_type(8)))  float    v8f;
typedef __attribute__((ext_vector_type(4)))  int      v4i;

// Problem constants (match reference)
#define Bsz 2
#define Nseq 1024
#define Dmodel 1024
#define Hheads 16
#define DRdim 128
// Derived
#define M1 2048      // B*N rows of stage-1 GEMM
#define K1 2048      // re||im concat of D
#define N1 8192      // [Qre | Qim | Kre | Kim] * (H*DR)
#define K2 256       // re||im concat of DR
#define BH 32        // B*H

// ---------- CDNA5 async global->LDS path (guarded; falls back to sync copies) ----------
#if defined(__HIP_DEVICE_COMPILE__) && \
    __has_builtin(__builtin_amdgcn_global_load_async_to_lds_b128) && \
    __has_builtin(__builtin_amdgcn_s_wait_asynccnt)
#define USE_ASYNC 1
#else
#define USE_ASYNC 0
#endif

#if USE_ASYNC
__device__ __forceinline__ void async_cp16(const f16* g, f16* l) {
    // GLOBAL_LOAD_ASYNC_TO_LDS_B128: 16 bytes/lane, tracked by ASYNCcnt
    // signature (from clang diagnostic): (as1 v4i*, as3 v4i*, imm offset, imm cpol)
    __builtin_amdgcn_global_load_async_to_lds_b128(
        (__attribute__((address_space(1))) v4i*)(g),
        (__attribute__((address_space(3))) v4i*)(l),
        0, 0);
}
#define WAIT_ASYNC(n) __builtin_amdgcn_s_wait_asynccnt(n)
#else
#define WAIT_ASYNC(n)
#endif

// ---------- WMMA fragment loaders (wave32, 16x16x32 f16) ----------
// A (16xK=32, f16): lane L -> row L&15; hi=L>>4 selects K offset 8.
// halves 0..7  = K [hi*8 .. hi*8+7], halves 8..15 = K [16+hi*8 .. 16+hi*8+7]
__device__ __forceinline__ v16h frag_a(const f16* __restrict__ tile, int ldh, int lane) {
    const int r = lane & 15, hi = lane >> 4;
    const f16* p = tile + r * ldh + hi * 8;
    v8h lo = *(const v8h*)(p);
    v8h hh = *(const v8h*)(p + 16);
    v16h f;
#pragma unroll
    for (int i = 0; i < 8; ++i) { f[i] = lo[i]; f[i + 8] = hh[i]; }
    return f;
}
// B (K=32 x 16, f16) stored as N x K rows (column n's k-vector contiguous):
// lane L -> col L&15; halves 0..7 = K [hi*8 .. hi*8+7], halves 8..15 = K [16+hi*8 .. 16+hi*8+7].
__device__ __forceinline__ v16h frag_b(const f16* __restrict__ tile, int ldh, int lane) {
    const int c = lane & 15, hi = lane >> 4;
    const f16* p = tile + c * ldh + hi * 8;
    v8h lo = *(const v8h*)(p);
    v8h hh = *(const v8h*)(p + 16);
    v16h f;
#pragma unroll
    for (int i = 0; i < 8; ++i) { f[i] = lo[i]; f[i + 8] = hh[i]; }
    return f;
}

// ---------- stage 0: f32 -> f16 operand builders ----------
__global__ __launch_bounds__(256) void convA(const float* __restrict__ xre,
                                             const float* __restrict__ xim,
                                             f16* __restrict__ Ah) {
    const unsigned i = blockIdx.x * 256u + threadIdx.x;   // < 2^22
    const int m = i >> 11, k = i & 2047;
    const float v = (k < Dmodel) ? xre[m * Dmodel + k] : xim[m * Dmodel + (k - Dmodel)];
    Ah[i] = (f16)v;
}
// Bh: 4 stacked (2048 x 2048) matrices, row = output column, k fastest.
// t=0: Qre = [Wq_re | -Wq_im]; t=1: Qim = [Wq_im | Wq_re]
// t=2: Kre = [Wk_re | -Wk_im]; t=3: Kim = [Wk_im | Wk_re]
__global__ __launch_bounds__(256) void convB(const float* __restrict__ qre,
                                             const float* __restrict__ qim,
                                             const float* __restrict__ kre,
                                             const float* __restrict__ kim,
                                             f16* __restrict__ Bh) {
    const unsigned i = blockIdx.x * 256u + threadIdx.x;   // < 2^24
    const int t = i >> 22;
    const int r = (i >> 11) & 2047;
    const int k = i & 2047;
    const int widx = r * Dmodel + (k & 1023);
    const bool hiK = k >= Dmodel;
    float v;
    switch (t) {
        case 0:  v = hiK ? -qim[widx] : qre[widx]; break;
        case 1:  v = hiK ?  qre[widx] : qim[widx]; break;
        case 2:  v = hiK ? -kim[widx] : kre[widx]; break;
        default: v = hiK ?  kre[widx] : kim[widx]; break;
    }
    Bh[i] = (f16)v;
}

// ---------- stage 1: C(2048x8192) = Ah @ Bh^T via WMMA, scatter to packed Q/K ----------
// 256 thr = 8 waves (2x4), block tile 128x256, wave tile 64x64, k-step 32,
// double-buffered LDS fed by async global->LDS copies.
__global__ __launch_bounds__(256) void s1_gemm(const f16* __restrict__ Ah,
                                               const f16* __restrict__ Bh,
                                               f16* __restrict__ Qp,
                                               f16* __restrict__ Kp) {
    __shared__ f16 As[2][128 * 32];
    __shared__ f16 Bs[2][256 * 32];
    const int lane = threadIdx.x & 31, wave = threadIdx.x >> 5;
    const int wr = wave >> 2, wc = wave & 3;
    const int rowBase = blockIdx.y * 128;
    const int colBase = blockIdx.x * 256;

    // per-thread copy slots
    const int ar = threadIdx.x >> 1;            // A: 128 rows x 2 half-rows
    const int ac = (threadIdx.x & 1) * 16;
    const int br = threadIdx.x;                 // B: 256 rows
    const f16* gA = Ah + (size_t)(rowBase + ar) * K1 + ac;
    const f16* gB = Bh + (size_t)(colBase + br) * K1;

    auto issue_tile = [&](int buf, int k0) {
#if USE_ASYNC
        async_cp16(gA + k0,      &As[buf][ar * 32 + ac]);
        async_cp16(gA + k0 + 8,  &As[buf][ar * 32 + ac + 8]);
        async_cp16(gB + k0,      &Bs[buf][br * 32]);
        async_cp16(gB + k0 + 8,  &Bs[buf][br * 32 + 8]);
        async_cp16(gB + k0 + 16, &Bs[buf][br * 32 + 16]);
        async_cp16(gB + k0 + 24, &Bs[buf][br * 32 + 24]);
#else
        *(v8h*)&As[buf][ar * 32 + ac]     = *(const v8h*)(gA + k0);
        *(v8h*)&As[buf][ar * 32 + ac + 8] = *(const v8h*)(gA + k0 + 8);
        *(v8h*)&Bs[buf][br * 32]          = *(const v8h*)(gB + k0);
        *(v8h*)&Bs[buf][br * 32 + 8]      = *(const v8h*)(gB + k0 + 8);
        *(v8h*)&Bs[buf][br * 32 + 16]     = *(const v8h*)(gB + k0 + 16);
        *(v8h*)&Bs[buf][br * 32 + 24]     = *(const v8h*)(gB + k0 + 24);
#endif
    };

    v8f acc[16] = {};
    const int T = K1 / 32;
    issue_tile(0, 0);
    for (int it = 0; it < T; ++it) {
        __syncthreads();                       // all waves done reading buf[(it+1)&1]
        if (it + 1 < T) {
            issue_tile((it + 1) & 1, (it + 1) * 32);
            WAIT_ASYNC(6);                     // 6 newer outstanding => tile(it) resident
        } else {
            WAIT_ASYNC(0);
        }
        __syncthreads();                       // tile(it) visible to all waves
        const f16* Ab = As[it & 1];
        const f16* Bb = Bs[it & 1];
        v16h afr[4];
#pragma unroll
        for (int i = 0; i < 4; ++i)
            afr[i] = frag_a(&Ab[(wr * 64 + i * 16) * 32], 32, lane);
#pragma unroll
        for (int j = 0; j < 4; ++j) {
            const v16h bfr = frag_b(&Bb[(wc * 64 + j * 16) * 32], 32, lane);
#pragma unroll
            for (int i = 0; i < 4; ++i)
                acc[i * 4 + j] = __builtin_amdgcn_wmma_f32_16x16x32_f16(
                    false, afr[i], false, bfr, (short)0, acc[i * 4 + j], false, false);
        }
    }
    // Epilogue: scatter f16 into packed (b,h,n, d + 128*isim) for Q and K
    const int nloc = lane & 15;
    const int rsel = (lane >> 4) * 8;
#pragma unroll
    for (int i = 0; i < 4; ++i) {
#pragma unroll
        for (int j = 0; j < 4; ++j) {
            const int colg = colBase + wc * 64 + j * 16 + nloc;
            const int t  = colg >> 11;
            const int c0 = colg & 2047;
            const int h  = c0 >> 7;
            const int d  = (c0 & 127) + ((t & 1) << 7);
            f16* __restrict__ dst = (t < 2) ? Qp : Kp;
            const v8f a = acc[i * 4 + j];
#pragma unroll
            for (int vg = 0; vg < 8; ++vg) {
                const int mg = rowBase + wr * 64 + i * 16 + rsel + vg;
                const int bq = mg >> 10, n = mg & 1023;
                dst[((size_t)((bq * Hheads + h) * Nseq + n)) * K2 + d] = (f16)a[vg];
            }
        }
    }
}

// ---------- stage 1.5: per-row unit L2 normalization (256 halves per row) ----------
__global__ __launch_bounds__(256) void norm_rows(f16* __restrict__ Qp, f16* __restrict__ Kp) {
    const int gwave = blockIdx.x * 8 + (threadIdx.x >> 5);   // 65536 waves total
    const int lane = threadIdx.x & 31;
    f16* __restrict__ buf = (gwave < BH * Nseq) ? Qp : Kp;
    const int row = gwave & (BH * Nseq - 1);
    f16* p = buf + (size_t)row * K2 + lane * 8;
    v8h v = *(const v8h*)p;
    float s = 0.f;
#pragma unroll
    for (int i = 0; i < 8; ++i) { const float f = (float)v[i]; s += f * f; }
#pragma unroll
    for (int ofs = 16; ofs > 0; ofs >>= 1) s += __shfl_xor(s, ofs, 32);
    const float inv = rsqrtf(s + 1e-12f);
#pragma unroll
    for (int i = 0; i < 8; ++i) v[i] = (f16)((float)v[i] * inv);
    *(v8h*)p = v;
}

// ---------- stage 2: per-(b,h) scores + sigmoid gate ----------
// S = Q(1024x256) @ K^T, inter = S * 2/128, out = sigmoid((inter - tau_d)/(1+1e-6))
__global__ __launch_bounds__(256) void s2_gemm(const f16* __restrict__ Qp,
                                               const f16* __restrict__ Kp,
                                               const float* __restrict__ wd,
                                               float* __restrict__ out) {
    __shared__ f16 As[2][128 * 32];
    __shared__ f16 Bs[2][256 * 32];
    const int bh = blockIdx.z;
    const f16* __restrict__ Qb = Qp + (size_t)bh * Nseq * K2;
    const f16* __restrict__ Kb = Kp + (size_t)bh * Nseq * K2;
    const int lane = threadIdx.x & 31, wave = threadIdx.x >> 5;
    const int wr = wave >> 2, wc = wave & 3;
    const int rowBase = blockIdx.y * 128;
    const int colBase = blockIdx.x * 256;

    const float w = wd[0];
    const float tau = -((w > 20.f) ? w : log1pf(__expf(w)));   // -softplus(w)
    const float invT = 1.0f / (1.0f + 1e-6f);

    const int ar = threadIdx.x >> 1;
    const int ac = (threadIdx.x & 1) * 16;
    const int br = threadIdx.x;
    const f16* gA = Qb + (size_t)(rowBase + ar) * K2 + ac;
    const f16* gB = Kb + (size_t)(colBase + br) * K2;

    auto issue_tile = [&](int buf, int k0) {
#if USE_ASYNC
        async_cp16(gA + k0,      &As[buf][ar * 32 + ac]);
        async_cp16(gA + k0 + 8,  &As[buf][ar * 32 + ac + 8]);
        async_cp16(gB + k0,      &Bs[buf][br * 32]);
        async_cp16(gB + k0 + 8,  &Bs[buf][br * 32 + 8]);
        async_cp16(gB + k0 + 16, &Bs[buf][br * 32 + 16]);
        async_cp16(gB + k0 + 24, &Bs[buf][br * 32 + 24]);
#else
        *(v8h*)&As[buf][ar * 32 + ac]     = *(const v8h*)(gA + k0);
        *(v8h*)&As[buf][ar * 32 + ac + 8] = *(const v8h*)(gA + k0 + 8);
        *(v8h*)&Bs[buf][br * 32]          = *(const v8h*)(gB + k0);
        *(v8h*)&Bs[buf][br * 32 + 8]      = *(const v8h*)(gB + k0 + 8);
        *(v8h*)&Bs[buf][br * 32 + 16]     = *(const v8h*)(gB + k0 + 16);
        *(v8h*)&Bs[buf][br * 32 + 24]     = *(const v8h*)(gB + k0 + 24);
#endif
    };

    v8f acc[16] = {};
    const int T = K2 / 32;
    issue_tile(0, 0);
    for (int it = 0; it < T; ++it) {
        __syncthreads();
        if (it + 1 < T) {
            issue_tile((it + 1) & 1, (it + 1) * 32);
            WAIT_ASYNC(6);
        } else {
            WAIT_ASYNC(0);
        }
        __syncthreads();
        const f16* Ab = As[it & 1];
        const f16* Bb = Bs[it & 1];
        v16h afr[4];
#pragma unroll
        for (int i = 0; i < 4; ++i)
            afr[i] = frag_a(&Ab[(wr * 64 + i * 16) * 32], 32, lane);
#pragma unroll
        for (int j = 0; j < 4; ++j) {
            const v16h bfr = frag_b(&Bb[(wc * 64 + j * 16) * 32], 32, lane);
#pragma unroll
            for (int i = 0; i < 4; ++i)
                acc[i * 4 + j] = __builtin_amdgcn_wmma_f32_16x16x32_f16(
                    false, afr[i], false, bfr, (short)0, acc[i * 4 + j], false, false);
        }
    }
    // Fused epilogue: scale, gate, store f32
    const int nloc = lane & 15;
    const int rsel = (lane >> 4) * 8;
    float* __restrict__ ob = out + (size_t)bh * Nseq * Nseq;
#pragma unroll
    for (int i = 0; i < 4; ++i) {
#pragma unroll
        for (int j = 0; j < 4; ++j) {
            const int colg = colBase + wc * 64 + j * 16 + nloc;
            const v8f a = acc[i * 4 + j];
#pragma unroll
            for (int vg = 0; vg < 8; ++vg) {
                const int mg = rowBase + wr * 64 + i * 16 + rsel + vg;
                const float inter = a[vg] * (2.0f / (float)DRdim);
                const float g = (inter - tau) * invT;
                ob[(size_t)mg * Nseq + colg] = 1.0f / (1.0f + __expf(-g));
            }
        }
    }
}

// ---------- launcher ----------
extern "C" void kernel_launch(void* const* d_in, const int* in_sizes, int n_in,
                              void* d_out, int out_size, void* d_ws, size_t ws_size,
                              hipStream_t stream) {
    (void)in_sizes; (void)n_in; (void)out_size; (void)ws_size;
    const float* xre = (const float*)d_in[0];
    const float* xim = (const float*)d_in[1];
    const float* wqr = (const float*)d_in[2];
    const float* wqi = (const float*)d_in[3];
    const float* wkr = (const float*)d_in[4];
    const float* wki = (const float*)d_in[5];
    const float* wd  = (const float*)d_in[6];

    // Workspace layout (total 75,497,472 bytes)
    char* w = (char*)d_ws;
    f16* Ah = (f16*)(w);                                    //  8 MiB: 2048x2048
    f16* Bh = (f16*)(w + (size_t)8388608);                  // 32 MiB: 8192x2048
    f16* Qp = (f16*)(w + (size_t)8388608 + 33554432);       // 16 MiB: 32x1024x256
    f16* Kp = (f16*)(w + (size_t)8388608 + 33554432 + 16777216);

    convA<<<(M1 * K1) / 256, 256, 0, stream>>>(xre, xim, Ah);
    convB<<<(N1 * K1) / 256, 256, 0, stream>>>(wqr, wqi, wkr, wki, Bh);

    dim3 g1(N1 / 256, M1 / 128, 1);
    s1_gemm<<<g1, 256, 0, stream>>>(Ah, Bh, Qp, Kp);

    norm_rows<<<(2 * BH * Nseq) / 8, 256, 0, stream>>>(Qp, Kp);

    dim3 g2(Nseq / 256, Nseq / 128, BH);
    s2_gemm<<<g2, 256, 0, stream>>>(Qp, Kp, wd, (float*)d_out);
}
